// RoPEMultiheadAttention_67181878444153
// MI455X (gfx1250) — matrix operand order repaired and checked by reading, not compiled, not run on hardware
//
#include <hip/hip_runtime.h>

// ---------- types ----------
typedef __attribute__((ext_vector_type(16))) __bf16         v16bf;
typedef __attribute__((ext_vector_type(8)))  float          v8f;
typedef __attribute__((ext_vector_type(4)))  float          f32x4;
typedef __attribute__((ext_vector_type(4)))  unsigned short u16x4;
typedef __attribute__((ext_vector_type(8)))  unsigned short u16x8;
typedef __attribute__((ext_vector_type(16))) unsigned short u16x16;

#define D_MODEL  1024
#define N_HEADS  16
#define HEAD_DIM 64
#define BATCH    2
#define SEQ      2048

// fp32 -> bf16 (RNE)
__device__ __forceinline__ unsigned short f2bf(float f) {
    unsigned int u = __builtin_bit_cast(unsigned int, f);
    u += 0x7FFFu + ((u >> 16) & 1u);
    return (unsigned short)(u >> 16);
}
__device__ __forceinline__ float bf2f(unsigned short u) {
    unsigned int v = ((unsigned int)u) << 16;
    return __builtin_bit_cast(float, v);
}
__device__ __forceinline__ v16bf mk16(u16x8 lo, u16x8 hi) {
    u16x16 t;
#pragma unroll
    for (int i = 0; i < 8; i++) { t[i] = lo[i]; t[8 + i] = hi[i]; }
    return __builtin_bit_cast(v16bf, t);
}

// CDNA5 async copy: 16B per lane, global -> LDS, tracked on ASYNCcnt
__device__ __forceinline__ void async_cp16(unsigned lds_off, const void* gaddr) {
    asm volatile("global_load_async_to_lds_b128 %0, %1, off"
                 :: "v"(lds_off), "v"(gaddr) : "memory");
}
__device__ __forceinline__ unsigned lds_addr(const void* p) {
    return (unsigned)(unsigned long long)p;   // low 32 bits of generic ptr = LDS offset
}

// =====================================================================
// elementwise fp32 -> bf16
// =====================================================================
__global__ void cvt_to_bf16(const float* __restrict__ s, unsigned short* __restrict__ d, int n) {
    int i = blockIdx.x * blockDim.x + threadIdx.x;
    if (i * 4 >= n) return;
    f32x4 v = *(const f32x4*)(s + i * 4);
    u16x4 o;
#pragma unroll
    for (int k = 0; k < 4; k++) o[k] = f2bf(v[k]);
    *(u16x4*)(d + i * 4) = o;
}

// =====================================================================
// GEMM: Y[M,1024] = X[M,1024] @ W[1024,1024]^T + bias   (X,W bf16)
// block 256 thr (8 waves), block tile 128(M)x64(N); wave tile 32x32
// double-buffered async staging; LDS rows padded to 72 (no bank conflicts
// among the 16 fragment rows: 144B stride -> first collision at dRow=16).
// =====================================================================
template<bool OUT_BF16>
__global__ void __launch_bounds__(256)
gemm_bf16(const unsigned short* __restrict__ X, const unsigned short* __restrict__ W,
          const float* __restrict__ bias, void* __restrict__ Yv)
{
    __shared__ __align__(16) unsigned short bufA[2][128 * 72];  // 2 x 18KB
    __shared__ __align__(16) unsigned short bufB[2][64 * 72];   // 2 x 9KB

    const int tid  = threadIdx.x;
    const int lane = tid & 31, wave = tid >> 5;
    const int half = lane >> 4, l16 = lane & 15;
    const int wm = wave & 3, wn = wave >> 2;
    const int mbase = blockIdx.y * 128;
    const int nbase = blockIdx.x * 64;

    const v8f vz = {0.f,0.f,0.f,0.f,0.f,0.f,0.f,0.f};
    v8f acc[2][2] = {{vz, vz}, {vz, vz}};

    auto stage = [&](int buf, int k0) {
        // A tile: 128 rows x 64 k = 1024 x 16B chunks
#pragma unroll
        for (int i = 0; i < 4; i++) {
            int c = tid + i * 256;
            int row = c >> 3, c8 = c & 7;
            async_cp16(lds_addr(&bufA[buf][row * 72 + c8 * 8]),
                       X + (size_t)(mbase + row) * D_MODEL + k0 + c8 * 8);
        }
        // B tile: 64 rows (n) x 64 k = 512 x 16B chunks
#pragma unroll
        for (int i = 0; i < 2; i++) {
            int c = tid + i * 256;
            int row = c >> 3, c8 = c & 7;
            async_cp16(lds_addr(&bufB[buf][row * 72 + c8 * 8]),
                       W + (size_t)(nbase + row) * D_MODEL + k0 + c8 * 8);
        }
    };

    stage(0, 0);
    const int NK = D_MODEL / 64;
    for (int it = 0; it < NK; ++it) {
        if (it + 1 < NK) {
            stage((it + 1) & 1, (it + 1) * 64);
            asm volatile("s_wait_asynccnt 0x6" ::: "memory");  // older 6 done (in-order)
        } else {
            asm volatile("s_wait_asynccnt 0x0" ::: "memory");
        }
        __syncthreads();

        const unsigned short* A  = bufA[it & 1];
        const unsigned short* Bt = bufB[it & 1];
#pragma unroll
        for (int ks = 0; ks < 2; ks++) {
            v16bf a[2], b[2];
#pragma unroll
            for (int mf = 0; mf < 2; mf++) {
                const unsigned short* p = &A[(wm * 32 + mf * 16 + l16) * 72 + ks * 32];
                a[mf] = mk16(*(const u16x8*)&p[8 * half], *(const u16x8*)&p[16 + 8 * half]);
            }
#pragma unroll
            for (int nf = 0; nf < 2; nf++) {
                const unsigned short* p = &Bt[(wn * 32 + nf * 16 + l16) * 72 + ks * 32];
                b[nf] = mk16(*(const u16x8*)&p[8 * half], *(const u16x8*)&p[16 + 8 * half]);
            }
#pragma unroll
            for (int mf = 0; mf < 2; mf++)
#pragma unroll
                for (int nf = 0; nf < 2; nf++)
                    acc[mf][nf] = __builtin_amdgcn_wmma_f32_16x16x32_bf16(
                        false, a[mf], false, b[nf], (short)0, acc[mf][nf], false, false);
        }
        __syncthreads();
    }

#pragma unroll
    for (int nf = 0; nf < 2; nf++) {
        int n = nbase + wn * 32 + nf * 16 + l16;
        float bv = bias[n];
#pragma unroll
        for (int mf = 0; mf < 2; mf++) {
#pragma unroll
            for (int r = 0; r < 8; r++) {
                int m = mbase + wm * 32 + mf * 16 + r + 8 * half;
                float v = acc[mf][nf][r] + bv;
                if (OUT_BF16) ((unsigned short*)Yv)[(size_t)m * D_MODEL + n] = f2bf(v);
                else          ((float*)Yv)[(size_t)m * D_MODEL + n] = v;
            }
        }
    }
}

// =====================================================================
// RoPE on Q,K + relayout [B,T,D] bf16 -> [B,H,T,hd] bf16 (V: relayout)
// =====================================================================
__global__ void rope_relayout(const unsigned short* __restrict__ Qg,
                              const unsigned short* __restrict__ Kg,
                              const unsigned short* __restrict__ Vg,
                              unsigned short* __restrict__ Qh,
                              unsigned short* __restrict__ Kh,
                              unsigned short* __restrict__ Vh)
{
    int gid = blockIdx.x * blockDim.x + threadIdx.x;   // B*T*H*32
    int j = gid & 31;
    int h = (gid >> 5) & 15;
    int t = (gid >> 9) & 2047;
    int b = gid >> 20;

    float invf = __expf(-(float)j * 0.03125f * 9.210340371976184f); // 10000^(-2j/64)
    float ang = (float)t * invf;
    float c = __cosf(ang), s = __sinf(ang);

    size_t src = ((size_t)(b * SEQ + t)) * D_MODEL + h * HEAD_DIM;
    size_t dst = (((size_t)(b * N_HEADS + h)) * SEQ + t) * HEAD_DIM;

    float q1 = bf2f(Qg[src + j]), q2 = bf2f(Qg[src + 32 + j]);
    Qh[dst + j]      = f2bf(q1 * c - q2 * s);
    Qh[dst + 32 + j] = f2bf(q2 * c + q1 * s);

    float k1 = bf2f(Kg[src + j]), k2 = bf2f(Kg[src + 32 + j]);
    Kh[dst + j]      = f2bf(k1 * c - k2 * s);
    Kh[dst + 32 + j] = f2bf(k2 * c + k1 * s);

    Vh[dst + j]      = Vg[src + j];
    Vh[dst + 32 + j] = Vg[src + 32 + j];
}

// =====================================================================
// Flash attention per (b,h): 128 thr (4 waves), wave = 16 query rows.
// 32-key chunks double-buffered on ASYNCcnt. Shuffle-free softmax:
// score tile bounced through LDS so each lane owns one full row.
// =====================================================================
__global__ void __launch_bounds__(128)
flash_attn(const unsigned short* __restrict__ Qh, const unsigned short* __restrict__ Kh,
           const unsigned short* __restrict__ Vh, unsigned short* __restrict__ Out)
{
    __shared__ __align__(16) unsigned short Kt[2][32 * 72];  // [key][dim] padded
    __shared__ __align__(16) unsigned short Vr[2][32 * 64];  // row-major staging
    __shared__ __align__(16) unsigned short Vt[64 * 40];     // [dim][key] padded
    __shared__ __align__(16) float          Sf[4][16 * 36];  // per-wave scores
    __shared__ __align__(16) unsigned short Pl[4][16 * 40];  // per-wave P (A layout)
    __shared__ __align__(16) float          Al[4][16];       // per-wave alpha / l

    const int tid  = threadIdx.x, lane = tid & 31, wave = tid >> 5;
    const int half = lane >> 4, l16 = lane & 15;
    const int bh = blockIdx.y;
    const int b = bh >> 4, h = bh & 15;
    const int q0 = blockIdx.x * 64 + wave * 16;
    const size_t headBase = (size_t)bh * SEQ * HEAD_DIM;

    // Q fragments (two 16x32 A frags over hd=64)
    const unsigned short* qrow = Qh + headBase + (size_t)(q0 + l16) * HEAD_DIM;
    v16bf qa[2];
#pragma unroll
    for (int f = 0; f < 2; f++) {
        u16x8 lo = *(const u16x8*)&qrow[f * 32 + 8 * half];
        u16x8 hi = *(const u16x8*)&qrow[f * 32 + 16 + 8 * half];
        qa[f] = mk16(lo, hi);
    }

    const v8f vz = {0.f,0.f,0.f,0.f,0.f,0.f,0.f,0.f};
    float m_run = -1e30f, l_run = 0.f;   // stats for row l16 (both halves redundant)
    v8f o[4];
#pragma unroll
    for (int g = 0; g < 4; g++) o[g] = vz;

    auto stageKV = [&](int buf, int kc) {
        const unsigned short* ks = Kh + headBase + (size_t)kc * HEAD_DIM;
        const unsigned short* vs = Vh + headBase + (size_t)kc * HEAD_DIM;
#pragma unroll
        for (int i = 0; i < 2; i++) {                 // 256 x 16B chunks
            int c = tid + i * 128;
            int row = c >> 3, c8 = c & 7;
            async_cp16(lds_addr(&Kt[buf][row * 72 + c8 * 8]),
                       ks + row * HEAD_DIM + c8 * 8);
        }
#pragma unroll
        for (int i = 0; i < 2; i++) {
            int c = tid + i * 128;
            int row = c >> 3, c8 = c & 7;
            async_cp16(lds_addr(&Vr[buf][c * 8]),
                       vs + row * HEAD_DIM + c8 * 8);
        }
    };

    stageKV(0, 0);
    const int NC = SEQ / 32;
    for (int it = 0; it < NC; ++it) {
        if (it + 1 < NC) {
            stageKV((it + 1) & 1, (it + 1) * 32);
            asm volatile("s_wait_asynccnt 0x4" ::: "memory");
        } else {
            asm volatile("s_wait_asynccnt 0x0" ::: "memory");
        }
        __syncthreads();
        const int cur = it & 1;

        // transpose Vr[cur] -> Vt (b128 LDS reads, b16 writes)
#pragma unroll
        for (int i = 0; i < 2; i++) {
            int c = tid + i * 128;
            int key = c >> 3, d8 = (c & 7) * 8;
            u16x8 w = *(const u16x8*)&Vr[cur][key * 64 + d8];
#pragma unroll
            for (int j = 0; j < 8; j++) Vt[(d8 + j) * 40 + key] = w[j];
        }

        // S = Q K^T (two 16x16 tiles over 32 keys)
        v8f s[2]; s[0] = vz; s[1] = vz;
#pragma unroll
        for (int nh = 0; nh < 2; nh++) {
#pragma unroll
            for (int ks2 = 0; ks2 < 2; ks2++) {
                const unsigned short* kp = &Kt[cur][(nh * 16 + l16) * 72 + ks2 * 32];
                v16bf kb = mk16(*(const u16x8*)&kp[8 * half], *(const u16x8*)&kp[16 + 8 * half]);
                s[nh] = __builtin_amdgcn_wmma_f32_16x16x32_bf16(
                    false, qa[ks2], false, kb, (short)0, s[nh], false, false);
            }
        }

        // bounce scaled scores so lane l16 owns full row l16
#pragma unroll
        for (int r = 0; r < 8; r++) {
            int row = r + 8 * half;
            Sf[wave][row * 36 + l16]      = s[0][r] * 0.125f;
            Sf[wave][row * 36 + 16 + l16] = s[1][r] * 0.125f;
        }
        asm volatile("s_wait_dscnt 0x0" ::: "memory");

        float p[32];
#pragma unroll
        for (int i = 0; i < 8; i++) {
            f32x4 v = *(const f32x4*)&Sf[wave][l16 * 36 + i * 4];
            p[i*4+0] = v[0]; p[i*4+1] = v[1]; p[i*4+2] = v[2]; p[i*4+3] = v[3];
        }
        float mx = p[0];
#pragma unroll
        for (int i = 1; i < 32; i++) mx = fmaxf(mx, p[i]);
        float mn = fmaxf(m_run, mx);
        float alpha = __expf(m_run - mn);
        float rs = 0.f;
#pragma unroll
        for (int i = 0; i < 32; i++) { p[i] = __expf(p[i] - mn); rs += p[i]; }
        l_run = l_run * alpha + rs;
        m_run = mn;

        if (half == 0) Al[wave][l16] = alpha;
        // store my half-row of P in A-fragment layout (2 x b128)
        u16x8 w0, w1;
#pragma unroll
        for (int j = 0; j < 8; j++) { w0[j] = f2bf(p[half*16 + j]); w1[j] = f2bf(p[half*16 + 8 + j]); }
        *(u16x8*)&Pl[wave][l16 * 40 + half * 16]     = w0;
        *(u16x8*)&Pl[wave][l16 * 40 + half * 16 + 8] = w1;
        asm volatile("s_wait_dscnt 0x0" ::: "memory");

        // rescale O rows (contiguous alpha read for rows 8*half..8*half+7)
        f32x4 a0 = *(const f32x4*)&Al[wave][8 * half];
        f32x4 a1 = *(const f32x4*)&Al[wave][8 * half + 4];
        float al[8] = {a0[0],a0[1],a0[2],a0[3],a1[0],a1[1],a1[2],a1[3]};
#pragma unroll
        for (int g = 0; g < 4; g++)
#pragma unroll
            for (int r = 0; r < 8; r++) o[g][r] *= al[r];

        const unsigned short* pp = &Pl[wave][l16 * 40];
        v16bf pa = mk16(*(const u16x8*)&pp[8 * half], *(const u16x8*)&pp[16 + 8 * half]);

        __syncthreads();   // Vt fully written across block
#pragma unroll
        for (int g = 0; g < 4; g++) {
            const unsigned short* vp = &Vt[(g * 16 + l16) * 40];
            v16bf vb = mk16(*(const u16x8*)&vp[8 * half], *(const u16x8*)&vp[16 + 8 * half]);
            o[g] = __builtin_amdgcn_wmma_f32_16x16x32_bf16(
                false, pa, false, vb, (short)0, o[g], false, false);
        }
        __syncthreads();   // done with Kt[cur]/Vt before next restage
    }

    // final normalization: publish l, read my rows' l contiguously
    if (half == 0) Al[wave][l16] = l_run;
    asm volatile("s_wait_dscnt 0x0" ::: "memory");
    f32x4 L0 = *(const f32x4*)&Al[wave][8 * half];
    f32x4 L1 = *(const f32x4*)&Al[wave][8 * half + 4];
    float li[8] = {L0[0],L0[1],L0[2],L0[3],L1[0],L1[1],L1[2],L1[3]};
#pragma unroll
    for (int r = 0; r < 8; r++) {
        float inv = 1.f / li[r];
        int row = q0 + r + 8 * half;
        size_t base = ((size_t)(b * SEQ + row)) * D_MODEL + h * HEAD_DIM;
#pragma unroll
        for (int g = 0; g < 4; g++)
            Out[base + g * 16 + l16] = f2bf(o[g][r] * inv);
    }
}

// =====================================================================
extern "C" void kernel_launch(void* const* d_in, const int* in_sizes, int n_in,
                              void* d_out, int out_size, void* d_ws, size_t ws_size,
                              hipStream_t stream)
{
    (void)in_sizes; (void)n_in; (void)out_size; (void)ws_size;

    const float* q_in = (const float*)d_in[0];
    const float* k_in = (const float*)d_in[1];
    const float* v_in = (const float*)d_in[2];
    const float* Wq   = (const float*)d_in[3];
    const float* bq   = (const float*)d_in[4];
    const float* Wk   = (const float*)d_in[5];
    const float* bk   = (const float*)d_in[6];
    const float* Wv   = (const float*)d_in[7];
    const float* bv   = (const float*)d_in[8];
    const float* Wo   = (const float*)d_in[9];
    const float* bo   = (const float*)d_in[10];
    float* out = (float*)d_out;

    const size_t eBTD = (size_t)BATCH * SEQ * D_MODEL;   // 4,194,304
    const size_t eDD  = (size_t)D_MODEL * D_MODEL;       // 1,048,576
    unsigned short* Xq  = (unsigned short*)d_ws;
    unsigned short* Xk  = Xq + eBTD;
    unsigned short* Xv  = Xk + eBTD;
    unsigned short* Qg  = Xv + eBTD;
    unsigned short* Kg  = Qg + eBTD;
    unsigned short* Vg  = Kg + eBTD;
    unsigned short* Wqh = Vg + eBTD;
    unsigned short* Wkh = Wqh + eDD;
    unsigned short* Wvh = Wkh + eDD;
    unsigned short* Woh = Wvh + eDD;
    unsigned short* Qhh = Xq;   // reuse after consumption
    unsigned short* Khh = Xk;
    unsigned short* Vhh = Xv;
    unsigned short* Ao  = Qg;

    cvt_to_bf16<<<(int)(eBTD / 1024), 256, 0, stream>>>(q_in, Xq, (int)eBTD);
    cvt_to_bf16<<<(int)(eBTD / 1024), 256, 0, stream>>>(k_in, Xk, (int)eBTD);
    cvt_to_bf16<<<(int)(eBTD / 1024), 256, 0, stream>>>(v_in, Xv, (int)eBTD);
    cvt_to_bf16<<<(int)(eDD  / 1024), 256, 0, stream>>>(Wq, Wqh, (int)eDD);
    cvt_to_bf16<<<(int)(eDD  / 1024), 256, 0, stream>>>(Wk, Wkh, (int)eDD);
    cvt_to_bf16<<<(int)(eDD  / 1024), 256, 0, stream>>>(Wv, Wvh, (int)eDD);
    cvt_to_bf16<<<(int)(eDD  / 1024), 256, 0, stream>>>(Wo, Woh, (int)eDD);

    dim3 ggrid(D_MODEL / 64, (BATCH * SEQ) / 128);
    gemm_bf16<true><<<ggrid, 256, 0, stream>>>(Xq, Wqh, bq, Qg);
    gemm_bf16<true><<<ggrid, 256, 0, stream>>>(Xk, Wkh, bk, Kg);
    gemm_bf16<true><<<ggrid, 256, 0, stream>>>(Xv, Wvh, bv, Vg);

    int total = BATCH * SEQ * N_HEADS * 32;
    rope_relayout<<<total / 256, 256, 0, stream>>>(Qg, Kg, Vg, Qhh, Khh, Vhh);

    dim3 agrid(SEQ / 64, BATCH * N_HEADS);
    flash_attn<<<agrid, 128, 0, stream>>>(Qhh, Khh, Vhh, Ao);

    gemm_bf16<false><<<ggrid, 256, 0, stream>>>(Ao, Woh, bo, out);
}
